// ProjectionLoss_5892695130344
// MI455X (gfx1250) — compile-verified, matrix operand order checked by reading, not run on hardware
//
#include <hip/hip_runtime.h>
#include <hip/hip_bf16.h>
#include <stdint.h>

// ---------------------------------------------------------------------------
// ProjectionLoss (Sinkhorn OT + alignment MSE) for MI455X / gfx1250.
//
// Strategy (see analysis): K = exp(-cdist(X,Y)/0.1) is materialized ONCE in
// bf16 (8192x8192x2B = 128 MB -> resident in the 192 MB global L2), produced
// by a bf16 WMMA GEMM with fused sqrt/exp epilogue. The 100 Sinkhorn
// iterations then stream K out of L2 (row matvec + split-K column matvec).
// The final <uKCv> contraction recomputes C tiles with a second WMMA pass.
// ---------------------------------------------------------------------------

typedef __attribute__((ext_vector_type(16))) __bf16 v16bf;
typedef __attribute__((ext_vector_type(8)))  float  v8f;

#define NROWS 8192
#define MCOLS 8192
#define DDIM  128
#define REG_INV 10.0f   // 1/0.1

union BF16Frag { uint32_t u[8]; v16bf v; };

__device__ __forceinline__ uint16_t f2bf(float f) {
    uint32_t x = __float_as_uint(f);
    uint32_t r = x + 0x7fffu + ((x >> 16) & 1u);   // round-to-nearest-even
    return (uint16_t)(r >> 16);
}
__device__ __forceinline__ float bflo(uint32_t x) { return __uint_as_float(x << 16); }
__device__ __forceinline__ float bfhi(uint32_t x) { return __uint_as_float(x & 0xffff0000u); }

__device__ __forceinline__ float wave_sum(float s) {
    #pragma unroll
    for (int m = 16; m > 0; m >>= 1) s += __shfl_xor(s, m, 32);
    return s;
}

// -------------------------- setup kernels ----------------------------------

__global__ void init_zero_kernel(float* __restrict__ KTu, float* __restrict__ scal) {
    int idx = blockIdx.x * 256 + threadIdx.x;
    KTu[idx] = 0.0f;
    if (idx < 4) scal[idx] = 0.0f;   // [0]=sum_tgt [1]=ot [2]=dist [3]=cnt
}

// Row norms + f32->bf16 conversion of X and Y (one wave per row).
__global__ void prep_kernel(const float* __restrict__ X, const float* __restrict__ Y,
                            float* __restrict__ xnorm, float* __restrict__ ynorm,
                            uint16_t* __restrict__ Xb, uint16_t* __restrict__ Yb) {
    int lane = threadIdx.x & 31;
    int row  = blockIdx.x * 8 + (threadIdx.x >> 5);

    float4 xv = *(const float4*)(X + (size_t)row * DDIM + lane * 4);
    float s = wave_sum(xv.x * xv.x + xv.y * xv.y + xv.z * xv.z + xv.w * xv.w);
    if (lane == 0) xnorm[row] = s;
    uint2 px;
    px.x = (uint32_t)f2bf(xv.x) | ((uint32_t)f2bf(xv.y) << 16);
    px.y = (uint32_t)f2bf(xv.z) | ((uint32_t)f2bf(xv.w) << 16);
    *(uint2*)(Xb + (size_t)row * DDIM + lane * 4) = px;

    float4 yv = *(const float4*)(Y + (size_t)row * DDIM + lane * 4);
    float t = wave_sum(yv.x * yv.x + yv.y * yv.y + yv.z * yv.z + yv.w * yv.w);
    if (lane == 0) ynorm[row] = t;
    uint2 py;
    py.x = (uint32_t)f2bf(yv.x) | ((uint32_t)f2bf(yv.y) << 16);
    py.y = (uint32_t)f2bf(yv.z) | ((uint32_t)f2bf(yv.w) << 16);
    *(uint2*)(Yb + (size_t)row * DDIM + lane * 4) = py;
}

__global__ void sum_tgt_kernel(const float* __restrict__ tgt, float* __restrict__ sum) {
    int idx = blockIdx.x * 256 + threadIdx.x;
    float s = wave_sum(tgt[idx]);
    if ((threadIdx.x & 31) == 0) atomicAdd(sum, s);
}

__global__ void uvb_init_kernel(const float* __restrict__ tgt, const float* __restrict__ scal,
                                float* __restrict__ b, float* __restrict__ u,
                                float* __restrict__ v) {
    int idx = blockIdx.x * 256 + threadIdx.x;
    b[idx] = tgt[idx] / scal[0];
    u[idx] = 1.0f / (float)NROWS;
    v[idx] = 1.0f / (float)MCOLS;
}

// ------------------ WMMA tile kernel (GEMM + epilogue) ---------------------
// STORE_K=true : write K = exp(-10*C) as bf16.
// STORE_K=false: accumulate ot += u_i * K_ij * C_ij * v_j (final einsum).
template <bool STORE_K>
__global__ __launch_bounds__(256)
void gemm_tiles_kernel(const uint16_t* __restrict__ Xb, const uint16_t* __restrict__ Yb,
                       const float* __restrict__ xnorm, const float* __restrict__ ynorm,
                       const float* __restrict__ u, const float* __restrict__ v,
                       uint16_t* __restrict__ Kout, float* __restrict__ ot_acc) {
    int lane = threadIdx.x & 31;
    int wid  = threadIdx.x >> 5;
    int wr = wid & 3;        // 4 row sub-blocks of 32
    int wc = wid >> 2;       // 2 col sub-blocks of 64
    int i0 = blockIdx.y * 128 + wr * 32;
    int j0 = blockIdx.x * 128 + wc * 64;

    v8f acc[2][4];
    #pragma unroll
    for (int t = 0; t < 2; ++t)
        #pragma unroll
        for (int s = 0; s < 4; ++s)
            acc[t][s] = (v8f){0.f, 0.f, 0.f, 0.f, 0.f, 0.f, 0.f, 0.f};

    int rrow  = lane & 15;          // row (A) / column (B) within 16-tile
    int khalf = (lane >> 4) * 8;    // lanes 16..31 hold the +8 K offset

    #pragma unroll
    for (int k0 = 0; k0 < DDIM; k0 += 32) {
        BF16Frag fa[2], fb[4];
        #pragma unroll
        for (int t = 0; t < 2; ++t) {
            const uint16_t* p = Xb + (size_t)(i0 + t * 16 + rrow) * DDIM + k0 + khalf;
            uint4 lo = *(const uint4*)p;
            uint4 hi = *(const uint4*)(p + 16);   // K += 16 elements
            fa[t].u[0] = lo.x; fa[t].u[1] = lo.y; fa[t].u[2] = lo.z; fa[t].u[3] = lo.w;
            fa[t].u[4] = hi.x; fa[t].u[5] = hi.y; fa[t].u[6] = hi.z; fa[t].u[7] = hi.w;
        }
        #pragma unroll
        for (int s = 0; s < 4; ++s) {
            const uint16_t* p = Yb + (size_t)(j0 + s * 16 + rrow) * DDIM + k0 + khalf;
            uint4 lo = *(const uint4*)p;
            uint4 hi = *(const uint4*)(p + 16);
            fb[s].u[0] = lo.x; fb[s].u[1] = lo.y; fb[s].u[2] = lo.z; fb[s].u[3] = lo.w;
            fb[s].u[4] = hi.x; fb[s].u[5] = hi.y; fb[s].u[6] = hi.z; fb[s].u[7] = hi.w;
        }
        #pragma unroll
        for (int t = 0; t < 2; ++t)
            #pragma unroll
            for (int s = 0; s < 4; ++s)
                acc[t][s] = __builtin_amdgcn_wmma_f32_16x16x32_bf16(
                    false, fa[t].v, false, fb[s].v, (short)0, acc[t][s], false, false);
    }

    // Epilogue: C/D layout -> i_local = e + 8*(lane>=16), j_local = lane&15.
    int ihalf = (lane >> 4) * 8;
    float lsum = 0.0f;
    #pragma unroll
    for (int t = 0; t < 2; ++t) {
        #pragma unroll
        for (int s = 0; s < 4; ++s) {
            #pragma unroll
            for (int e = 0; e < 8; ++e) {
                int i = i0 + t * 16 + ihalf + e;
                int j = j0 + s * 16 + (lane & 15);
                float dot = acc[t][s][e];
                float c2  = xnorm[i] + ynorm[j] - 2.0f * dot;
                float c   = sqrtf(fmaxf(c2, 0.0f));
                float kv  = __expf(-c * REG_INV);
                if (STORE_K) {
                    Kout[(size_t)i * MCOLS + j] = f2bf(kv);
                } else {
                    lsum += u[i] * kv * c * v[j];
                }
            }
        }
    }
    if (!STORE_K) {
        lsum = wave_sum(lsum);
        if (lane == 0) atomicAdd(ot_acc, lsum);
    }
}

// ------------------------- Sinkhorn matvecs --------------------------------

// u[i] = (1/N) / (K[i,:] . v)   -- one wave per 16KB row, 512B/wave loads.
__global__ __launch_bounds__(256)
void mv_rows_u_kernel(const uint16_t* __restrict__ Kb, const float* __restrict__ v,
                      float* __restrict__ u) {
    int lane = threadIdx.x & 31;
    int row  = blockIdx.x * 8 + (threadIdx.x >> 5);
    const uint16_t* kr = Kb + (size_t)row * MCOLS;
    float dot = 0.0f;
    #pragma unroll 4
    for (int j = lane * 8; j < MCOLS; j += 256) {
        uint4  kk = *(const uint4*)(kr + j);
        float4 v0 = *(const float4*)(v + j);
        float4 v1 = *(const float4*)(v + j + 4);
        dot += bflo(kk.x) * v0.x + bfhi(kk.x) * v0.y;
        dot += bflo(kk.y) * v0.z + bfhi(kk.y) * v0.w;
        dot += bflo(kk.z) * v1.x + bfhi(kk.z) * v1.y;
        dot += bflo(kk.w) * v1.z + bfhi(kk.w) * v1.w;
    }
    dot = wave_sum(dot);
    if (lane == 0) u[row] = (1.0f / (float)NROWS) / dot;
}

// KTu[j] += sum_i K[i,j]*u[i]  (split-K over grid.y for memory parallelism)
__global__ __launch_bounds__(256)
void mv_cols_partial_kernel(const uint16_t* __restrict__ Kb, const float* __restrict__ u,
                            float* __restrict__ KTu) {
    int col0 = (blockIdx.x * 256 + threadIdx.x) * 2;  // 2 adjacent cols per thread
    int i0   = blockIdx.y * 256;
    float a0 = 0.0f, a1 = 0.0f;
    #pragma unroll 8
    for (int i = i0; i < i0 + 256; ++i) {
        uint32_t kk = *(const uint32_t*)(Kb + (size_t)i * MCOLS + col0);
        float ui = u[i];                       // uniform -> scalar load
        a0 += bflo(kk) * ui;
        a1 += bfhi(kk) * ui;
    }
    atomicAdd(&KTu[col0],     a0);
    atomicAdd(&KTu[col0 + 1], a1);
}

// v[j] = b[j]/KTu[j], and re-zero KTu for the next iteration.
__global__ void v_update_kernel(const float* __restrict__ b, float* __restrict__ KTu,
                                float* __restrict__ v) {
    int j = blockIdx.x * 256 + threadIdx.x;
    v[j] = b[j] / KTu[j];
    KTu[j] = 0.0f;
}

// ------------------------- alignment MSE + finalize ------------------------

__global__ void dist_kernel(const float* __restrict__ X, const float* __restrict__ Y,
                            const int* __restrict__ aligned,
                            float* __restrict__ dist_acc, float* __restrict__ cnt_acc) {
    int lane = threadIdx.x & 31;
    int row  = blockIdx.x * 8 + (threadIdx.x >> 5);
    int al   = aligned[row];
    if (al < 0) return;    // uniform per wave
    float4 xv = *(const float4*)(X + (size_t)row * DDIM + lane * 4);
    float4 yv = *(const float4*)(Y + (size_t)al  * DDIM + lane * 4);
    float dx = xv.x - yv.x, dy = xv.y - yv.y, dz = xv.z - yv.z, dw = xv.w - yv.w;
    float s = wave_sum(dx * dx + dy * dy + dz * dz + dw * dw);
    if (lane == 0) {
        atomicAdd(dist_acc, s);
        atomicAdd(cnt_acc, 1.0f);
    }
}

__global__ void finalize_kernel(const float* __restrict__ scal, float* __restrict__ out) {
    if (threadIdx.x == 0 && blockIdx.x == 0) {
        float cnt = scal[3];
        float d   = (cnt > 0.0f) ? scal[2] / (cnt * (float)DDIM) : 0.0f;
        out[0] = scal[1] + d;
    }
}

// ------------------------------ launcher -----------------------------------

extern "C" void kernel_launch(void* const* d_in, const int* in_sizes, int n_in,
                              void* d_out, int out_size, void* d_ws, size_t ws_size,
                              hipStream_t stream) {
    (void)in_sizes; (void)n_in; (void)out_size; (void)ws_size;

    const float* X       = (const float*)d_in[0];   // descriptors (8192,128)
    const float* Y       = (const float*)d_in[1];   // word_embeddings (8192,128)
    const int*   aligned = (const int*)  d_in[2];   // (8192,)
    const float* tgt     = (const float*)d_in[3];   // (8192,)
    float* out = (float*)d_out;

    // Workspace layout (needs ~132 MB; K kept bf16 so it is L2-resident).
    char* ws = (char*)d_ws;
    size_t off = 0;
    uint16_t* Kb    = (uint16_t*)(ws + off); off += (size_t)NROWS * MCOLS * 2;  // 128 MB
    float*    xnorm = (float*)(ws + off);    off += (size_t)NROWS * 4;
    float*    ynorm = (float*)(ws + off);    off += (size_t)MCOLS * 4;
    float*    u     = (float*)(ws + off);    off += (size_t)NROWS * 4;
    float*    v     = (float*)(ws + off);    off += (size_t)MCOLS * 4;
    float*    b     = (float*)(ws + off);    off += (size_t)MCOLS * 4;
    float*    KTu   = (float*)(ws + off);    off += (size_t)MCOLS * 4;
    float*    scal  = (float*)(ws + off);    off += 4 * 4;                       // sum/ot/dist/cnt
    uint16_t* Xb    = (uint16_t*)(ws + off); off += (size_t)NROWS * DDIM * 2;
    uint16_t* Yb    = (uint16_t*)(ws + off); off += (size_t)MCOLS * DDIM * 2;

    // Setup
    init_zero_kernel<<<32, 256, 0, stream>>>(KTu, scal);
    prep_kernel<<<1024, 256, 0, stream>>>(X, Y, xnorm, ynorm, Xb, Yb);
    sum_tgt_kernel<<<32, 256, 0, stream>>>(tgt, &scal[0]);
    uvb_init_kernel<<<32, 256, 0, stream>>>(tgt, scal, b, u, v);

    // K = exp(-10 * cdist(X,Y))  [bf16, WMMA GEMM + fused epilogue]
    gemm_tiles_kernel<true><<<dim3(64, 64), 256, 0, stream>>>(
        Xb, Yb, xnorm, ynorm, nullptr, nullptr, Kb, nullptr);

    // Sinkhorn iterations (K streamed from L2)
    for (int it = 0; it < 100; ++it) {
        mv_rows_u_kernel<<<1024, 256, 0, stream>>>(Kb, v, u);
        mv_cols_partial_kernel<<<dim3(16, 32), 256, 0, stream>>>(Kb, u, KTu);
        v_update_kernel<<<32, 256, 0, stream>>>(b, KTu, v);
    }

    // ot = sum_ij u_i K_ij C_ij v_j  (recompute C tiles with WMMA)
    gemm_tiles_kernel<false><<<dim3(64, 64), 256, 0, stream>>>(
        Xb, Yb, xnorm, ynorm, u, v, nullptr, &scal[1]);

    // alignment MSE + combine
    dist_kernel<<<1024, 256, 0, stream>>>(X, Y, aligned, &scal[2], &scal[3]);
    finalize_kernel<<<1, 32, 0, stream>>>(scal, out);
}
